// ChebNetRouting2Layer_32229434589303
// MI455X (gfx1250) — matrix operand order repaired and checked by reading, not compiled, not run on hardware
//
#include <hip/hip_runtime.h>
#include <hip/hip_bf16.h>

// ---------------------------------------------------------------------------
// ChebNet 2-layer (K=3) for MI455X / gfx1250.
//  - Sparse propagation: coalesced float4 gather + global_atomic_add_f32 scatter
//  - Dense Tx_k @ W_k: V_WMMA_F32_16X16X32_F16 with split-f16 (hi+lo) operands
//    emulating f32 precision; all three Chebyshev terms + bias + ReLU fused.
// ---------------------------------------------------------------------------

typedef _Float16 v16h __attribute__((ext_vector_type(16)));
typedef float    v8f  __attribute__((ext_vector_type(8)));

// ----------------------------- utility kernels -----------------------------

__global__ __launch_bounds__(256) void zero_f32_kernel(float* __restrict__ p, long n) {
    long i = (long)blockIdx.x * blockDim.x + threadIdx.x;
    if (i < n) p[i] = 0.0f;
}

__global__ __launch_bounds__(256) void scale_copy_kernel(float* __restrict__ dst,
                                                         const float* __restrict__ src,
                                                         long n, float s) {
    long i = (long)blockIdx.x * blockDim.x + threadIdx.x;
    if (i < n) dst[i] = s * src[i];
}

// deg[row] += w[e]
__global__ __launch_bounds__(256) void scatter_deg_kernel(float* __restrict__ deg,
                                                          const long long* __restrict__ src,
                                                          const float* __restrict__ w, int E) {
    int e = blockIdx.x * blockDim.x + threadIdx.x;
    if (e < E) atomicAdd(&deg[(long)src[e]], w[e]);
}

// deg -> d^{-1/2} in place (0 if deg<=0)
__global__ __launch_bounds__(256) void dinv_kernel(float* __restrict__ deg, int N) {
    int i = blockIdx.x * blockDim.x + threadIdx.x;
    if (i < N) {
        float d = deg[i];
        deg[i] = (d > 0.0f) ? rsqrtf(d) : 0.0f;
    }
}

// norm[e] = -dinv[src] * w * dinv[dst]
__global__ __launch_bounds__(256) void norm_kernel(float* __restrict__ norm,
                                                   const float* __restrict__ dinv,
                                                   const long long* __restrict__ src,
                                                   const long long* __restrict__ dst,
                                                   const float* __restrict__ w, int E) {
    int e = blockIdx.x * blockDim.x + threadIdx.x;
    if (e < E) {
        norm[e] = -dinv[(long)src[e]] * w[e] * dinv[(long)dst[e]];
    }
}

// out[dst[e], :] += scale * norm[e] * x[src[e], :]   (F = FG*4 features)
// Consecutive threads cover one edge's feature row -> coalesced gather+scatter.
template <int FG>
__global__ __launch_bounds__(256) void scatter_feat_kernel(float* __restrict__ out,
                                                           const float* __restrict__ x,
                                                           const float* __restrict__ norm,
                                                           const long long* __restrict__ src,
                                                           const long long* __restrict__ dst,
                                                           int E, float scale) {
    long tid = (long)blockIdx.x * blockDim.x + threadIdx.x;
    int e = (int)(tid / FG);
    int g = (int)(tid % FG);
    if (e >= E) return;
    const long F = FG * 4;
    const float4* xr = reinterpret_cast<const float4*>(x + (long)src[e] * F);
    float4 v = xr[g];
    float s = scale * norm[e];
    float* o = out + (long)dst[e] * F + (long)g * 4;
    atomicAdd(o + 0, s * v.x);
    atomicAdd(o + 1, s * v.y);
    atomicAdd(o + 2, s * v.z);
    atomicAdd(o + 3, s * v.w);
}

// --------------------------- fused WMMA GEMM -------------------------------
// out[N,64] = act( bias + A0@W[0] + A1@W[1] + A2@W[2] ),  W: [3, F, 64], f32.
// f32 emulated via split-f16: A = Ah + Al, B = Bh + Bl (exact residual split),
// C += Ah*Bh + Ah*Bl + Al*Bh accumulated in f32 by V_WMMA_F32_16X16X32_F16.
// One wave = one 16x16 output tile; block = 128 threads = 4 waves = 64 cols.
template <int F>
__global__ __launch_bounds__(128) void cheb_gemm_wmma_kernel(
        const float* __restrict__ A0, const float* __restrict__ A1,
        const float* __restrict__ A2, const float* __restrict__ W,
        const float* __restrict__ bias, float* __restrict__ out, int relu) {
    constexpr int H = 64;
    const int lane   = threadIdx.x & 31;
    const int wave   = threadIdx.x >> 5;
    const int row0   = blockIdx.x * 16;
    const int hbase  = wave * 16;
    const int m      = lane & 15;        // A row / B,C col within tile
    const int lhalf  = lane >> 4;        // 0 or 1
    const int kbaseA = lhalf * 8;        // A: K offset per lane-half
    const int kbaseB = lhalf * 16;       // unused: B loads use kbaseA, same K order as A

    // C fragment init with bias (broadcast down rows)
    v8f c;
    const float bv = bias[hbase + m];
#pragma unroll
    for (int r = 0; r < 8; ++r) c[r] = bv;

    const float* As[3] = {A0, A1, A2};
#pragma unroll
    for (int t = 0; t < 3; ++t) {
        const float* __restrict__ arow = As[t] + (long)(row0 + m) * F;
        const float* __restrict__ Wt   = W + (long)t * F * H;
#pragma unroll
        for (int kk = 0; kk < F; kk += 32) {
            v16h ah, al, bh, bl;
            // A fragment: elements 0..7 -> K = kk+kbaseA+0..7,
            //             elements 8..15 -> K = kk+16+kbaseA+0..7
#pragma unroll
            for (int i = 0; i < 16; ++i) {
                int k = kk + ((i < 8) ? (kbaseA + i) : (16 + kbaseA + (i - 8)));
                float av = arow[k];
                _Float16 hi = (_Float16)av;
                ah[i] = hi;
                al[i] = (_Float16)(av - (float)hi);
            }
            // B fragment: same K order as A, col = hbase+m
#pragma unroll
            for (int i = 0; i < 16; ++i) {
                int k = kk + ((i < 8) ? (kbaseA + i) : (16 + kbaseA + (i - 8)));
                float wv = Wt[(long)k * H + hbase + m];
                _Float16 hi = (_Float16)wv;
                bh[i] = hi;
                bl[i] = (_Float16)(wv - (float)hi);
            }
            c = __builtin_amdgcn_wmma_f32_16x16x32_f16(false, ah, false, bh,
                                                       (short)0, c, false, false);
            c = __builtin_amdgcn_wmma_f32_16x16x32_f16(false, ah, false, bl,
                                                       (short)0, c, false, false);
            c = __builtin_amdgcn_wmma_f32_16x16x32_f16(false, al, false, bh,
                                                       (short)0, c, false, false);
        }
    }

    // C layout: VGPR r -> M = r + (lane>>4)*8, N = lane&15
#pragma unroll
    for (int r = 0; r < 8; ++r) {
        float v = c[r];
        if (relu) v = v > 0.0f ? v : 0.0f;
        out[(long)(row0 + r + lhalf * 8) * H + hbase + m] = v;
    }
}

// ------------------------------- launcher ----------------------------------

extern "C" void kernel_launch(void* const* d_in, const int* in_sizes, int n_in,
                              void* d_out, int out_size, void* d_ws, size_t ws_size,
                              hipStream_t stream) {
    (void)n_in; (void)out_size; (void)ws_size;

    const float*     x   = (const float*)d_in[0];
    const long long* ei  = (const long long*)d_in[1];   // int64 edge_index [2,E]
    const float*     w   = (const float*)d_in[2];
    const float*     W1  = (const float*)d_in[3];       // [3,32,64]
    const float*     b1  = (const float*)d_in[4];
    const float*     W2  = (const float*)d_in[5];       // [3,64,64]
    const float*     b2  = (const float*)d_in[6];

    const int IN = 32, H = 64;
    const int N = in_sizes[0] / IN;        // 100000
    const int E = in_sizes[2];             // 1600000
    const long long* src = ei;             // row
    const long long* dst = ei + E;         // col

    // Workspace carve-up (256B aligned): deg | norm | T1 | T2 | Hbuf
    char* ws = (char*)d_ws;
    auto al = [](size_t v) { return (v + 255) & ~(size_t)255; };
    size_t off = 0;
    float* deg  = (float*)(ws + off); off += al((size_t)N * 4);
    float* norm = (float*)(ws + off); off += al((size_t)E * 4);
    float* T1   = (float*)(ws + off); off += al((size_t)N * H * 4);
    float* T2   = (float*)(ws + off); off += al((size_t)N * H * 4);
    float* Hbuf = (float*)(ws + off); off += al((size_t)N * H * 4);

    const int TB = 256;
    auto blocks = [&](long n) { return (unsigned)((n + TB - 1) / TB); };

    // ---- symmetric normalization ----
    zero_f32_kernel<<<blocks(N), TB, 0, stream>>>(deg, N);
    scatter_deg_kernel<<<blocks(E), TB, 0, stream>>>(deg, src, w, E);
    dinv_kernel<<<blocks(N), TB, 0, stream>>>(deg, N);
    norm_kernel<<<blocks(E), TB, 0, stream>>>(norm, deg, src, dst, w, E);

    // ---- layer 1 (F = 32) ----
    zero_f32_kernel<<<blocks((long)N * IN), TB, 0, stream>>>(T1, (long)N * IN);
    scatter_feat_kernel<8><<<blocks((long)E * 8), TB, 0, stream>>>(T1, x, norm, src, dst, E, 1.0f);
    scale_copy_kernel<<<blocks((long)N * IN), TB, 0, stream>>>(T2, x, (long)N * IN, -1.0f);
    scatter_feat_kernel<8><<<blocks((long)E * 8), TB, 0, stream>>>(T2, T1, norm, src, dst, E, 2.0f);
    cheb_gemm_wmma_kernel<32><<<N / 16, 128, 0, stream>>>(x, T1, T2, W1, b1, Hbuf, /*relu=*/1);

    // ---- layer 2 (F = 64) ----
    zero_f32_kernel<<<blocks((long)N * H), TB, 0, stream>>>(T1, (long)N * H);
    scatter_feat_kernel<16><<<blocks((long)E * 16), TB, 0, stream>>>(T1, Hbuf, norm, src, dst, E, 1.0f);
    scale_copy_kernel<<<blocks((long)N * H), TB, 0, stream>>>(T2, Hbuf, (long)N * H, -1.0f);
    scatter_feat_kernel<16><<<blocks((long)E * 16), TB, 0, stream>>>(T2, T1, norm, src, dst, E, 2.0f);
    cheb_gemm_wmma_kernel<64><<<N / 16, 128, 0, stream>>>(Hbuf, T1, T2, W2, b2, (float*)d_out, /*relu=*/0);
}
